// LSTMModel_62216896250311
// MI455X (gfx1250) — matrix operand order repaired and checked by reading, not compiled, not run on hardware
//
#include <hip/hip_runtime.h>

typedef __attribute__((ext_vector_type(16))) _Float16 v16h;
typedef __attribute__((ext_vector_type(8)))  float    v8f;

#define T_LEN   512
#define D_IN    5
#define H_SZ    10
#define BATCH_N 8192

// SWAPX16 via ds_swizzle: offset = xor_mask(0x10)<<10 | and_mask(0x1f)
__device__ __forceinline__ float swz16(float x) {
  int i = __builtin_amdgcn_ds_swizzle(__float_as_int(x), 0x401F);
  return __int_as_float(i);
}
// sigmoid(x) = 1/(1+2^(-x*log2e)) via v_exp_f32 + v_rcp_f32
__device__ __forceinline__ float fast_sig(float x) {
  float e = __builtin_amdgcn_exp2f(-1.4426950408889634f * x);
  return __builtin_amdgcn_rcpf(1.0f + e);
}
// tanh(x) = 2*sigmoid(2x)-1
__device__ __forceinline__ float fast_tanh(float x) {
  float e = __builtin_amdgcn_exp2f(-2.8853900817779268f * x);
  return 2.0f * __builtin_amdgcn_rcpf(1.0f + e) - 1.0f;
}

__global__ __launch_bounds__(256, 1)
void lstm_wmma_kernel(const float* __restrict__ x,
                      const float* __restrict__ W_ih,
                      const float* __restrict__ W_hh,
                      const float* __restrict__ b_ih,
                      const float* __restrict__ b_hh,
                      const float* __restrict__ W_out,
                      const float* __restrict__ b_out,
                      float* __restrict__ out) {
  const int lane    = threadIdx.x & 31;
  const int waveId  = (int)((blockIdx.x * blockDim.x + threadIdx.x) >> 5);
  const int colBase = waveId * 16;          // 16 batch columns per wave
  const int n       = lane & 15;            // batch column within group
  const bool hiHalf = lane >= 16;
  const int m       = lane & 15;            // A-matrix row (gate-hidden idx)

  // ---- Build time-invariant A tiles (one 16x32 f16 tile per gate) ----
  // K layout: 0..4 = x, 5..14 = h, 15 = bias(+1 column), 16..31 = zero pad.
  // A k order used here (contiguous, same for B): lanes0-15 element e: K=e (0..15); lanes16-31: K=16+e (zero pad).
  v16h A[4];
  #pragma unroll
  for (int g = 0; g < 4; ++g) {
    #pragma unroll
    for (int e = 0; e < 16; ++e) A[g][e] = (_Float16)0.0f;
    const int row = g * H_SZ + m;           // row of [4H, *] weight matrices
    if (m < H_SZ && !hiHalf) {
      #pragma unroll
      for (int v = 0; v < 8; ++v) {
        #pragma unroll
        for (int s = 0; s < 2; ++s) {
          const int k = 2 * v + s;
          float wv;
          if (k < D_IN)                 wv = W_ih[row * D_IN + k];
          else if (k < D_IN + H_SZ)     wv = W_hh[row * H_SZ + (k - D_IN)];
          else                          wv = b_ih[row] + b_hh[row];   // k == 15
          A[g][2 * v + s] = (_Float16)wv;
        }
      }
    }
  }

  // ---- Recurrent state in D-matrix layout: elem (m = r + 16*hiHalf? no:
  // m = r + (hiHalf?8:0), n = lane&15). m in 10..15 are dead pad rows
  // (zero weights -> gates==0 -> state stays finite). ----
  float c[8], h[8];
  #pragma unroll
  for (int r = 0; r < 8; ++r) { c[r] = 0.0f; h[r] = 0.0f; }

  const float* xp = x + (size_t)(colBase + n) * (size_t)(T_LEN * D_IN);
  float xr[D_IN];
  #pragma unroll
  for (int d = 0; d < D_IN; ++d) xr[d] = xp[d];   // t = 0

  const v8f zacc = {0.f, 0.f, 0.f, 0.f, 0.f, 0.f, 0.f, 0.f};

  for (int t = 0; t < T_LEN; ++t) {
    // ---- Build B = [x_t ; h ; 1]^T (32x16 f16, K in elements 0..15) ----
    v16h Bm;
    Bm[0]  = (_Float16)xr[0];
    Bm[1]  = (_Float16)xr[1];
    Bm[2]  = (_Float16)xr[2];
    Bm[3]  = (_Float16)xr[3];
    Bm[4]  = (_Float16)xr[4];
    Bm[5]  = (_Float16)h[0];
    Bm[6]  = (_Float16)h[1];
    Bm[7]  = (_Float16)h[2];
    Bm[8]  = (_Float16)h[3];
    Bm[9]  = (_Float16)h[4];
    Bm[10] = (_Float16)h[5];
    Bm[11] = (_Float16)h[6];
    Bm[12] = (_Float16)h[7];
    Bm[13] = (_Float16)swz16(h[0]);   // h[8] lives in high lane-half of h[0]
    Bm[14] = (_Float16)swz16(h[1]);   // h[9] lives in high lane-half of h[1]
    Bm[15] = (_Float16)1.0f;          // bias column

    // ---- Gates via WMMA: D[m=gate-hidden, n=batch] ----
    v8f gi = __builtin_amdgcn_wmma_f32_16x16x32_f16(false, A[0], false, Bm, (short)0, zacc, false, false);
    v8f gf = __builtin_amdgcn_wmma_f32_16x16x32_f16(false, A[1], false, Bm, (short)0, zacc, false, false);
    v8f gg = __builtin_amdgcn_wmma_f32_16x16x32_f16(false, A[2], false, Bm, (short)0, zacc, false, false);
    v8f go = __builtin_amdgcn_wmma_f32_16x16x32_f16(false, A[3], false, Bm, (short)0, zacc, false, false);

    // ---- Prefetch next timestep's x while WMMA/activations run ----
    if (t + 1 < T_LEN) {
      const float* xn = xp + (size_t)(t + 1) * D_IN;
      if ((t & 15) == 0) __builtin_prefetch(xn + 16 * D_IN, 0, 1);
      #pragma unroll
      for (int d = 0; d < D_IN; ++d) xr[d] = xn[d];
    }

    // ---- Lane-local LSTM cell update (i/f/g/o/c/h all share one layout) ----
    #pragma unroll
    for (int r = 0; r < 8; ++r) {
      float ig = fast_sig(gi[r]);
      float fg = fast_sig(gf[r]);
      float gv = fast_tanh(gg[r]);
      float og = fast_sig(go[r]);
      float cn = fg * c[r] + ig * gv;
      c[r] = cn;
      h[r] = og * fast_tanh(cn);
    }
  }

  // ---- Output head: out[b] = sum_k h[k,b] * W_out[k] + b_out ----
  float part = 0.0f;
  #pragma unroll
  for (int r = 0; r < 8; ++r) {
    const int mr = r + (hiHalf ? 8 : 0);
    const float w = (mr < H_SZ) ? W_out[mr] : 0.0f;
    part = fmaf(h[r], w, part);
  }
  part += swz16(part);                 // combine the two lane-halves
  if (!hiHalf) out[colBase + n] = part + b_out[0];
}

extern "C" void kernel_launch(void* const* d_in, const int* in_sizes, int n_in,
                              void* d_out, int out_size, void* d_ws, size_t ws_size,
                              hipStream_t stream) {
  const float* x     = (const float*)d_in[0];
  const float* W_ih  = (const float*)d_in[1];
  const float* W_hh  = (const float*)d_in[2];
  const float* b_ih  = (const float*)d_in[3];
  const float* b_hh  = (const float*)d_in[4];
  const float* W_out = (const float*)d_in[5];
  const float* b_out = (const float*)d_in[6];
  float* out = (float*)d_out;

  // 512 waves total (one per 16 batch cols): 64 blocks x 256 threads (8 waves)
  dim3 grid(BATCH_N / (16 * 8));
  dim3 block(256);
  lstm_wmma_kernel<<<grid, block, 0, stream>>>(x, W_ih, W_hh, b_ih, b_hh,
                                               W_out, b_out, out);
}
